// loss_create_graph_2680059592917
// MI455X (gfx1250) — compile-verified, matrix operand order checked by reading, not run on hardware
//
#include <hip/hip_runtime.h>
#include <math.h>

// ---- problem constants (from reference) ----
constexpr int N = 8192;
constexpr int D = 128;
constexpr int TILE = 16;
constexpr int TJ = 8;                         // j-tiles per wave (amortize A-fragment loads)
constexpr int ITILES = N / TILE;              // 512
constexpr int JGROUPS = ITILES / TJ;          // 64
constexpr long long TOTAL_WAVES = (long long)ITILES * JGROUPS;  // 32768

typedef __attribute__((ext_vector_type(16))) _Float16 v16h;
typedef __attribute__((ext_vector_type(8)))  _Float16 v8h;
typedef __attribute__((ext_vector_type(8)))  float    v8f;

// ---- kernel 0: zero the global accumulator (ws is poisoned, must re-zero each call) ----
__global__ void k_zero(float* __restrict__ accum) {
    if (threadIdx.x == 0 && blockIdx.x == 0) accum[0] = 0.0f;
}

// ---- kernel 1: x (f32) -> xh (f16), row-major [N, D] ----
__global__ void k_cvt(const float* __restrict__ x, _Float16* __restrict__ xh) {
    int idx = blockIdx.x * blockDim.x + threadIdx.x;
    if (idx < N * D) xh[idx] = (_Float16)x[idx];
}

// Load one 16x32 (MxK) f16 fragment of X rows [row0..row0+15], K-block kb.
// ISA 16-bit A layout (mirrored for B with lane<->N):
//   lanes 0-15:  M = lane,    K = {0..7} in v[0..7], K = {16..23} in v[8..15]
//   lanes 16-31: M = lane-16, K = {8..15},           K = {24..31}
__device__ __forceinline__ v16h load_frag(const _Float16* __restrict__ xh,
                                          int row0, int kb, int laneN, int klo) {
    const _Float16* p = xh + (size_t)(row0 + laneN) * D + kb * 32 + klo;
    v8h lo = *(const v8h*)(p);        // K = klo .. klo+7
    v8h hi = *(const v8h*)(p + 16);   // K = klo+16 .. klo+23
    return __builtin_shufflevector(lo, hi, 0,1,2,3,4,5,6,7, 8,9,10,11,12,13,14,15);
}

// ---- kernel 2: fused  scores = Xh Xh^T (WMMA f16) -> double-sigmoid BCE -> global sum ----
__global__ void k_gemm_bce(const _Float16* __restrict__ xh,
                           const float* __restrict__ adj,
                           float* __restrict__ accum) {
    const int lane = threadIdx.x & 31;
    const long long wave = ((long long)blockIdx.x * blockDim.x + threadIdx.x) >> 5;
    if (wave >= TOTAL_WAVES) return;

    const int i_tile  = (int)(wave / JGROUPS);
    const int j_group = (int)(wave % JGROUPS);
    const int i0 = i_tile * TILE;

    const int laneN = lane & 15;
    const int klo   = (lane >> 4) ? 8 : 0;   // K sub-offset per half-wave
    const int mbase = (lane >> 4) ? 8 : 0;   // C-layout: lanes>=16 hold M = r+8

    // A fragments for the full K=128 (4 blocks of 32), loaded once per wave
    v16h a0 = load_frag(xh, i0, 0, laneN, klo);
    v16h a1 = load_frag(xh, i0, 1, laneN, klo);
    v16h a2 = load_frag(xh, i0, 2, laneN, klo);
    v16h a3 = load_frag(xh, i0, 3, laneN, klo);

    float lsum = 0.0f;

    for (int jt = 0; jt < TJ; ++jt) {
        const int j0 = (j_group * TJ + jt) * TILE;

        // Issue the adj loads (the only HBM-bound traffic) FIRST so they
        // overlap with the B-fragment loads (L2-resident) and the WMMA chain.
        const float* adjp = adj + (size_t)(i0 + mbase) * N + j0 + laneN;
        float av[8];
        #pragma unroll
        for (int r = 0; r < 8; ++r) av[r] = adjp[(size_t)r * N];

        // B fragments: rows j0..j0+15 of X, same packing (X * X^T symmetry)
        v16h b0 = load_frag(xh, j0, 0, laneN, klo);
        v16h b1 = load_frag(xh, j0, 1, laneN, klo);
        v16h b2 = load_frag(xh, j0, 2, laneN, klo);
        v16h b3 = load_frag(xh, j0, 3, laneN, klo);

        v8f c = {};
        c = __builtin_amdgcn_wmma_f32_16x16x32_f16(false, a0, false, b0, (short)0, c, false, false);
        c = __builtin_amdgcn_wmma_f32_16x16x32_f16(false, a1, false, b1, (short)0, c, false, false);
        c = __builtin_amdgcn_wmma_f32_16x16x32_f16(false, a2, false, b2, (short)0, c, false, false);
        c = __builtin_amdgcn_wmma_f32_16x16x32_f16(false, a3, false, b3, (short)0, c, false, false);

        // c[r] = scores[i0 + mbase + r][j0 + laneN]
        #pragma unroll
        for (int r = 0; r < 8; ++r) {
            float s = c[r];
            // z = sigmoid(s) via fast rcp (v_rcp_f32, no IEEE div expansion)
            float z  = __builtin_amdgcn_rcpf(1.0f + __expf(-s));
            // softplus(-z) = log(1 + e^{-z}); z in (0,1) -> fast log is safe
            float lp = __logf(1.0f + __expf(-z));
            // bce = z*(1-t) + lp ; t = (adj>0)
            lsum += (av[r] > 0.0f) ? lp : (z + lp);
        }
    }

    // wave32 reduction, then one atomic per wave
    #pragma unroll
    for (int off = 16; off > 0; off >>= 1)
        lsum += __shfl_xor(lsum, off, 32);
    if (lane == 0) atomicAdd(accum, lsum);
}

// ---- kernel 3: mean ----
__global__ void k_final(const float* __restrict__ accum, float* __restrict__ out) {
    if (threadIdx.x == 0 && blockIdx.x == 0)
        out[0] = accum[0] * (1.0f / ((float)N * (float)N));
}

extern "C" void kernel_launch(void* const* d_in, const int* in_sizes, int n_in,
                              void* d_out, int out_size, void* d_ws, size_t ws_size,
                              hipStream_t stream) {
    const float* x   = (const float*)d_in[0];   // [N, D] f32
    const float* adj = (const float*)d_in[1];   // [N, N] f32
    float* out = (float*)d_out;                 // scalar f32

    // workspace layout: [accum f32 @0 .. 256) | xh f16 [N*D] @256
    float*     accum = (float*)d_ws;
    _Float16*  xh    = (_Float16*)((char*)d_ws + 256);

    k_zero<<<1, 1, 0, stream>>>(accum);

    int cvt_threads = N * D;
    k_cvt<<<(cvt_threads + 255) / 256, 256, 0, stream>>>(x, xh);

    long long total_threads = TOTAL_WAVES * 32;            // 1,048,576
    int block = 256;                                       // 8 waves / block
    int grid  = (int)((total_threads + block - 1) / block);
    k_gemm_bce<<<grid, block, 0, stream>>>(xh, adj, accum);

    k_final<<<1, 1, 0, stream>>>(accum, out);
}
